// MultiHeadAttention_4922032521398
// MI455X (gfx1250) — matrix operand order repaired and checked by reading, not compiled, not run on hardware
//
#include <hip/hip_runtime.h>

typedef __bf16 bf16_t;
typedef __attribute__((ext_vector_type(16))) __bf16 v16bf;
typedef __attribute__((ext_vector_type(4)))  __bf16 v4bf;
typedef __attribute__((ext_vector_type(8)))  float  v8f;

#define B_ 32
#define L_ 512
#define DIM_ 256
#define NH_ 4
#define HD_ 64
#define NWORDS 16  /* 512 bits / 32 */

// ---- helpers -------------------------------------------------------------

__device__ inline bf16_t f2bf(float f) {
  unsigned u = __builtin_bit_cast(unsigned, f);
  unsigned r = u + 0x7FFFu + ((u >> 16) & 1u);   // round-to-nearest-even
  unsigned short h = (unsigned short)(r >> 16);
  return __builtin_bit_cast(bf16_t, h);
}

__device__ inline v8f wmma_bf16(v16bf a, v16bf b, v8f c) {
  return __builtin_amdgcn_wmma_f32_16x16x32_bf16(
      /*neg_a=*/false, a, /*neg_b=*/false, b,
      /*c_mod=*/(short)0, c, /*reuse_a=*/false, /*reuse_b=*/false);
}

// wave-local LDS ordering (wave-private LDS slice; no block barrier needed)
__device__ inline void wave_lds_fence() {
  __builtin_amdgcn_wave_barrier();
  asm volatile("s_wait_dscnt 0" ::: "memory");
  __builtin_amdgcn_wave_barrier();
}

// A fragment, 16x32 (MxK) bf16, per ISA 7.12.2: lane l holds row l%16,
// K = (l/16)*8 + e (e<8) and 16 + (l/16)*8 + (e-8) (e>=8). Source row-major.
__device__ inline v16bf load_a_frag(const bf16_t* base, int ld) {
  int lane = threadIdx.x & 31;
  const bf16_t* p = base + (lane & 15) * ld + (lane >> 4) * 8;
  v16bf r;
#pragma unroll
  for (int e = 0; e < 8; ++e) r[e] = p[e];
#pragma unroll
  for (int e = 0; e < 8; ++e) r[8 + e] = p[16 + e];
  return r;
}

// B fragment, 32x16 (KxN) bf16, where MEMORY is row-major [N][K] (i.e. we use
// it transposed: W[N,K], K-matrix rows, V^T rows). Lane l = column l%16;
// element e maps to K = (l/16)*8 + e (e<8), 16 + (l/16)*8 + (e-8) (e>=8).
__device__ inline v16bf load_b_fragT(const bf16_t* base, int ld) {
  int lane = threadIdx.x & 31;
  const bf16_t* p = base + (lane & 15) * ld + (lane >> 4) * 8;
  v16bf r;
#pragma unroll
  for (int e = 0; e < 8; ++e) r[e] = p[e];
#pragma unroll
  for (int e = 0; e < 8; ++e) r[8 + e] = p[16 + e];
  return r;
}

// ---- 0) one-shot fp32 -> bf16 conversion (removes VALU from GEMM loops) --

__global__ __launch_bounds__(256)
void cvt_kernel(const float* __restrict__ src, bf16_t* __restrict__ dst, int n4) {
  int t = blockIdx.x * blockDim.x + threadIdx.x;
  if (t >= n4) return;
  float4 v = ((const float4*)src)[t];
  v4bf r;
  r[0] = f2bf(v.x); r[1] = f2bf(v.y); r[2] = f2bf(v.z); r[3] = f2bf(v.w);
  ((v4bf*)dst)[t] = r;
}

// ---- 1) QKV projection: qkv = x @ w_qkv^T, scatter to q/k/vT bf16 --------
// One wave computes a 16x64 strip (4 N-tiles) so each A fragment feeds 4 WMMAs.

__global__ __launch_bounds__(256)
void qkv_gemm_kernel(const bf16_t* __restrict__ xb, const bf16_t* __restrict__ wqb,
                     bf16_t* __restrict__ qb, bf16_t* __restrict__ kb,
                     bf16_t* __restrict__ vb) {
  int wid = blockIdx.x * (blockDim.x >> 5) + (threadIdx.x >> 5);
  const int NQ = (3 * DIM_) / 64;          // 12 quad col-groups
  int nq = wid % NQ, mt = wid / NQ;        // mt < 1024
  if (mt >= (B_ * L_) / 16) return;
  int lane = threadIdx.x & 31;
  const bf16_t* arow = xb + (size_t)(mt * 16) * DIM_;
  const bf16_t* wrow = wqb + (size_t)(nq * 64) * DIM_;
  v8f acc[4] = {{}, {}, {}, {}};
#pragma unroll
  for (int kk = 0; kk < DIM_; kk += 32) {
    v16bf a = load_a_frag(arow + kk, DIM_);
#pragma unroll
    for (int t = 0; t < 4; ++t) {
      v16bf b = load_b_fragT(wrow + (size_t)(t * 16) * DIM_ + kk, DIM_);
      acc[t] = wmma_bf16(a, b, acc[t]);
    }
  }
  int half = lane >> 4, ln = lane & 15;
  // 64-col quad never crosses a q/k/v (256) or head (64) boundary
  int colbase = nq * 64;
  int sel = colbase >> 8;                  // 0=q 1=k 2=v (wave-uniform)
  int h = (colbase & 255) >> 6;            // head (wave-uniform)
#pragma unroll
  for (int t = 0; t < 4; ++t) {
    int d = t * 16 + ln;
#pragma unroll
    for (int jv = 0; jv < 8; ++jv) {
      int rowg = mt * 16 + jv + 8 * half;
      int bb = rowg >> 9, li = rowg & (L_ - 1);
      int bh = bb * NH_ + h;
      float v = acc[t][jv];
      if (sel == 0)      qb[((size_t)bh * L_ + li) * HD_ + d] = f2bf(v * 0.125f); // fold 1/sqrt(64)
      else if (sel == 1) kb[((size_t)bh * L_ + li) * HD_ + d] = f2bf(v);
      else               vb[((size_t)bh * HD_ + d) * L_ + li] = f2bf(v);          // V transposed
    }
  }
}

// ---- 2) binarize adjacency + bit-pack rows & columns ---------------------

__global__ __launch_bounds__(256)
void pack_bits_kernel(const int* __restrict__ adj, unsigned* __restrict__ rowbits,
                      unsigned* __restrict__ colbits) {
  int t = blockIdx.x * blockDim.x + threadIdx.x;
  const int N = B_ * L_ * NWORDS;          // 262144
  bool colmode = t >= N;
  int id = colmode ? t - N : t;
  if (id >= N) return;
  int w = id & (NWORDS - 1);
  int i = (id >> 4) & (L_ - 1);
  int b = id >> 13;
  const int* base = adj + (size_t)b * L_ * L_;
  unsigned word = 0;
#pragma unroll 4
  for (int tb = 0; tb < 32; ++tb) {
    int v = colmode ? base[(size_t)(w * 32 + tb) * L_ + i]
                    : base[(size_t)i * L_ + w * 32 + tb];
    int nz = (v != 0) && !((v > 1) && (v < 9));   // a = binarize(where(1<adj<9, 0, adj))
    word |= ((unsigned)nz) << tb;
  }
  (colmode ? colbits : rowbits)[id] = word;
}

// ---- 3) per-head packed masks: pe = [a, aT, aT@a, a@aT] + I, !=0 ---------

__global__ __launch_bounds__(256)
void mask_kernel(const unsigned* __restrict__ rowbits, const unsigned* __restrict__ colbits,
                 unsigned* __restrict__ pmask) {
  int t = blockIdx.x * blockDim.x + threadIdx.x;
  if (t >= B_ * NH_ * L_ * NWORDS) return;
  int w = t & 15;
  int i = (t >> 4) & (L_ - 1);
  int head = (t >> 13) & 3;
  int b = t >> 15;
  unsigned word = 0;
  if (head < 2) {
    const unsigned* src = (head == 0) ? rowbits : colbits;   // a, aT
    word = src[((size_t)b * L_ + i) * NWORDS + w];
  } else {
    // (aT@a)[i,j]!=0 <=> colbits[i]&colbits[j]!=0 ; (a@aT): rowbits version
    const unsigned* src = (head == 2) ? colbits : rowbits;
    const unsigned* ri = src + ((size_t)b * L_ + i) * NWORDS;
    unsigned mine[NWORDS];
#pragma unroll
    for (int q = 0; q < NWORDS; ++q) mine[q] = ri[q];
    for (int tb = 0; tb < 32; ++tb) {
      const unsigned* rj = src + ((size_t)b * L_ + (w * 32 + tb)) * NWORDS;
      unsigned acc = 0;
#pragma unroll
      for (int q = 0; q < NWORDS; ++q) acc |= mine[q] & rj[q];
      word |= (acc ? 1u : 0u) << tb;
    }
  }
  if ((i >> 5) == w) word |= 1u << (i & 31);   // + eye(L)
  pmask[((size_t)((b * NH_ + head) * L_) + i) * NWORDS + w] = word;
}

// ---- 4) flash attention: one wave = 16 query rows x full head ------------

__global__ __launch_bounds__(128)
void attn_kernel(const bf16_t* __restrict__ qb, const bf16_t* __restrict__ kb,
                 const bf16_t* __restrict__ vb, const unsigned* __restrict__ pmask,
                 bf16_t* __restrict__ attnbuf) {
  __shared__ bf16_t plds[4][16 * 32];      // per-wave P staging tile
  int wave = threadIdx.x >> 5;
  int lane = threadIdx.x & 31;
  int half = lane >> 4, ln = lane & 15;
  int gw = blockIdx.x * 4 + wave;          // 0..4095
  int qt = gw & 31;                        // query tile (16 rows)
  int bh = gw >> 5;                        // b*NH + h
  int h = bh & (NH_ - 1), b = bh >> 2;
  const bf16_t* qbase = qb + ((size_t)bh * L_ + qt * 16) * HD_;
  const bf16_t* kbase = kb + (size_t)bh * L_ * HD_;
  const bf16_t* vbase = vb + (size_t)bh * HD_ * L_;   // [HD][L]
  const unsigned* mbase = pmask + ((size_t)bh * L_ + qt * 16) * NWORDS;
  bf16_t* P = &plds[wave][0];

  v16bf aq0 = load_a_frag(qbase, HD_);       // d 0..31 (already scaled)
  v16bf aq1 = load_a_frag(qbase + 32, HD_);  // d 32..63
  float m[8], lsum[8];
  v8f o0 = {}, o1 = {}, o2 = {}, o3 = {};
#pragma unroll
  for (int jv = 0; jv < 8; ++jv) { m[jv] = -1e30f; lsum[jv] = 0.f; }

  for (int kbk = 0; kbk < L_ / 32; ++kbk) {
    int key0 = kbk * 32;
    // scores for 32 keys: S = q . k^T  (k rows contiguous over d)
    v8f slo = {}, shi = {};
    {
      v16bf b0 = load_b_fragT(kbase + (size_t)key0 * HD_, HD_);
      v16bf b1 = load_b_fragT(kbase + (size_t)key0 * HD_ + 32, HD_);
      slo = wmma_bf16(aq0, b0, slo);
      slo = wmma_bf16(aq1, b1, slo);
      v16bf c0 = load_b_fragT(kbase + (size_t)(key0 + 16) * HD_, HD_);
      v16bf c1 = load_b_fragT(kbase + (size_t)(key0 + 16) * HD_ + 32, HD_);
      shi = wmma_bf16(aq0, c0, shi);
      shi = wmma_bf16(aq1, c1, shi);
    }
    // masked online softmax; row r = jv + 8*half lives across lanes 0-15 / 16-31
#pragma unroll
    for (int jv = 0; jv < 8; ++jv) {
      unsigned wword = mbase[(size_t)(jv + 8 * half) * NWORDS + kbk];
      int blo = (wword >> ln) & 1;
      int bhi = (wword >> (16 + ln)) & 1;
      float sl = blo ? slo[jv] : -1e30f;
      float sh = bhi ? shi[jv] : -1e30f;
      float tmax = fmaxf(sl, sh);
#pragma unroll
      for (int mk = 1; mk < 16; mk <<= 1) tmax = fmaxf(tmax, __shfl_xor(tmax, mk, 32));
      float nm = fmaxf(m[jv], tmax);
      float fac = __expf(m[jv] - nm);
      m[jv] = nm;
      float pl = blo ? __expf(slo[jv] - nm) : 0.f;   // explicit zero for masked
      float ph = bhi ? __expf(shi[jv] - nm) : 0.f;
      float rs = pl + ph;
#pragma unroll
      for (int mk = 1; mk < 16; mk <<= 1) rs += __shfl_xor(rs, mk, 32);
      lsum[jv] = lsum[jv] * fac + rs;
      o0[jv] *= fac; o1[jv] *= fac; o2[jv] *= fac; o3[jv] *= fac;
      P[(jv + 8 * half) * 32 + ln] = f2bf(pl);
      P[(jv + 8 * half) * 32 + 16 + ln] = f2bf(ph);
    }
    wave_lds_fence();
    v16bf ap = load_a_frag(P, 32);           // P as 16x32 A fragment
    {
      v16bf bv0 = load_b_fragT(vbase + (size_t)0 * 16 * L_ + key0, L_);
      o0 = wmma_bf16(ap, bv0, o0);
      v16bf bv1 = load_b_fragT(vbase + (size_t)1 * 16 * L_ + key0, L_);
      o1 = wmma_bf16(ap, bv1, o1);
      v16bf bv2 = load_b_fragT(vbase + (size_t)2 * 16 * L_ + key0, L_);
      o2 = wmma_bf16(ap, bv2, o2);
      v16bf bv3 = load_b_fragT(vbase + (size_t)3 * 16 * L_ + key0, L_);
      o3 = wmma_bf16(ap, bv3, o3);
    }
    wave_lds_fence();                        // keep P reads ahead of next writes
  }
#pragma unroll
  for (int jv = 0; jv < 8; ++jv) {
    float inv = 1.0f / lsum[jv];             // diag bit guarantees lsum > 0
    int li = qt * 16 + jv + 8 * half;
    bf16_t* orow = attnbuf + ((size_t)b * L_ + li) * DIM_ + h * HD_;
    orow[0 * 16 + ln] = f2bf(o0[jv] * inv);
    orow[1 * 16 + ln] = f2bf(o1[jv] * inv);
    orow[2 * 16 + ln] = f2bf(o2[jv] * inv);
    orow[3 * 16 + ln] = f2bf(o3[jv] * inv);
  }
}

// ---- 5) output projection: out = attn @ w_proj^T (fp32 out) --------------
// One wave computes a 16x64 strip (4 N-tiles).

__global__ __launch_bounds__(256)
void proj_gemm_kernel(const bf16_t* __restrict__ attnbuf, const bf16_t* __restrict__ wpb,
                      float* __restrict__ out) {
  int wid = blockIdx.x * (blockDim.x >> 5) + (threadIdx.x >> 5);
  const int NQ = DIM_ / 64;                // 4 quad col-groups
  int nq = wid & (NQ - 1), mt = wid / NQ;
  if (mt >= (B_ * L_) / 16) return;
  int lane = threadIdx.x & 31;
  const bf16_t* arow = attnbuf + (size_t)(mt * 16) * DIM_;
  const bf16_t* wrow = wpb + (size_t)(nq * 64) * DIM_;
  v8f acc[4] = {{}, {}, {}, {}};
#pragma unroll
  for (int kk = 0; kk < DIM_; kk += 32) {
    v16bf a = load_a_frag(arow + kk, DIM_);
#pragma unroll
    for (int t = 0; t < 4; ++t) {
      v16bf b = load_b_fragT(wrow + (size_t)(t * 16) * DIM_ + kk, DIM_);
      acc[t] = wmma_bf16(a, b, acc[t]);
    }
  }
  int half = lane >> 4, ln = lane & 15;
#pragma unroll
  for (int t = 0; t < 4; ++t) {
#pragma unroll
    for (int jv = 0; jv < 8; ++jv) {
      int rowg = mt * 16 + jv + 8 * half;
      out[(size_t)rowg * DIM_ + nq * 64 + t * 16 + ln] = acc[t][jv];
    }
  }
}

// ---- launcher ------------------------------------------------------------

extern "C" void kernel_launch(void* const* d_in, const int* in_sizes, int n_in,
                              void* d_out, int out_size, void* d_ws, size_t ws_size,
                              hipStream_t stream) {
  const float* x     = (const float*)d_in[0];
  const int*   adj   = (const int*)d_in[1];
  const float* wqkv  = (const float*)d_in[2];
  const float* wproj = (const float*)d_in[3];
  float* out = (float*)d_out;

  char* p = (char*)d_ws;
  const size_t QK = (size_t)B_ * NH_ * L_ * HD_ * sizeof(bf16_t);   // 8 MB
  bf16_t* qb = (bf16_t*)p;        p += QK;
  bf16_t* kb = (bf16_t*)p;        p += QK;
  bf16_t* vb = (bf16_t*)p;        p += QK;
  unsigned* rowbits = (unsigned*)p; p += (size_t)B_ * L_ * NWORDS * 4;
  unsigned* colbits = (unsigned*)p; p += (size_t)B_ * L_ * NWORDS * 4;
  unsigned* pmask   = (unsigned*)p; p += (size_t)B_ * NH_ * L_ * NWORDS * 4;
  bf16_t* attnbuf = (bf16_t*)p;   p += (size_t)B_ * L_ * DIM_ * sizeof(bf16_t);
  bf16_t* xb  = (bf16_t*)p;       p += (size_t)B_ * L_ * DIM_ * sizeof(bf16_t);
  bf16_t* wqb = (bf16_t*)p;       p += (size_t)3 * DIM_ * DIM_ * sizeof(bf16_t);
  bf16_t* wpb = (bf16_t*)p;       // DIM*DIM bf16

  {  // 0) bf16 pre-conversion of GEMM operands
    int n4 = (B_ * L_ * DIM_) / 4;
    cvt_kernel<<<(n4 + 255) / 256, 256, 0, stream>>>(x, xb, n4);
    n4 = (3 * DIM_ * DIM_) / 4;
    cvt_kernel<<<(n4 + 255) / 256, 256, 0, stream>>>(wqkv, wqb, n4);
    n4 = (DIM_ * DIM_) / 4;
    cvt_kernel<<<(n4 + 255) / 256, 256, 0, stream>>>(wproj, wpb, n4);
  }
  {  // 1) QKV GEMM: 1024 x 12 quad tiles, 8 waves/block
    int waves = ((B_ * L_) / 16) * ((3 * DIM_) / 64);
    qkv_gemm_kernel<<<waves / 8, 256, 0, stream>>>(xb, wqb, qb, kb, vb);
  }
  {  // 2) bit-pack adjacency
    int items = 2 * B_ * L_ * NWORDS;
    pack_bits_kernel<<<(items + 255) / 256, 256, 0, stream>>>(adj, rowbits, colbits);
  }
  {  // 3) per-head packed masks
    int items = B_ * NH_ * L_ * NWORDS;
    mask_kernel<<<(items + 255) / 256, 256, 0, stream>>>(rowbits, colbits, pmask);
  }
  {  // 4) flash attention: 4096 waves, 4 waves/block
    int waves = B_ * NH_ * (L_ / 16);
    attn_kernel<<<waves / 4, 128, 0, stream>>>(qb, kb, vb, pmask, attnbuf);
  }
  {  // 5) projection GEMM: 1024 x 4 quad tiles, 8 waves/block
    int waves = ((B_ * L_) / 16) * (DIM_ / 64);
    proj_gemm_kernel<<<waves / 8, 256, 0, stream>>>(attnbuf, wpb, out);
  }
}
